// TitleEncoder_12627203850732
// MI455X (gfx1250) — matrix operand order repaired and checked by reading, not compiled, not run on hardware
//
#include <hip/hip_runtime.h>

// ---------- types ----------
typedef __attribute__((ext_vector_type(16))) __bf16 v16bf;
typedef __attribute__((ext_vector_type(8)))  __bf16 v8bf;
typedef __attribute__((ext_vector_type(8)))  float  v8f;

#define B_TOTAL   32768
#define N_WORD    30
#define WORD_DIM  300
#define KPAD_W    320      // word-dim padded to 10 * 32
#define CHANNEL   400      // 25 tiles of 16
#define CPAD      416      // channel padded to 13 * 32 (attn K)
#define ATTN_DIM  200
#define APAD      208      // attn dim padded to 13 * 16

// workspace layout (bytes, all 256-aligned)
#define WS_A_OFF     0u
#define WS_A_BYTES   (B_TOTAL * N_WORD * 4u)            // 3,932,160
#define WS_WC_OFF    (WS_A_OFF + WS_A_BYTES)
#define WS_WC_BYTES  768256u                            // 750 blocks * 512 bf16 (+pad)
#define WS_WV_OFF    (WS_WC_OFF + WS_WC_BYTES)
#define WS_WV_BYTES  173056u                            // 169 blocks * 512 bf16
#define WS_ST_OFF    (WS_WV_OFF + WS_WV_BYTES)

__device__ __forceinline__ __bf16 f2bf(float f) {
    unsigned u = __float_as_uint(f);
    unsigned r = (u + 0x7FFFu + ((u >> 16) & 1u)) >> 16;   // round-to-nearest-even
    unsigned short s = (unsigned short)r;
    __bf16 out;
    __builtin_memcpy(&out, &s, 2);
    return out;
}

// A-fragment (bf16 16xK slice, K-chunk kc of 32) from LDS, row stride in elements.
// ISA layout: lanes 0-15 hold M=lane, K {0..7,16..23}; lanes 16-31 same M, +8.
__device__ __forceinline__ v16bf load_a(const __bf16* base, int stride, int kc) {
    int l = threadIdx.x & 31;
    int m = l & 15;
    int off = (l & 16) ? 8 : 0;
    const __bf16* p = base + m * stride + kc * 32 + off;
    union { v16bf v; v8bf h[2]; } u;
    u.h[0] = *(const v8bf*)(p);        // K off..off+7
    u.h[1] = *(const v8bf*)(p + 16);   // K off+16..off+23
    return u.v;
}

// B-fragment: pre-swizzled block of 512 bf16; lane reads 16 contiguous values.
__device__ __forceinline__ v16bf load_b(const __bf16* block) {
    int l = threadIdx.x & 31;
    return *(const v16bf*)(block + l * 16);
}

// Fill one 16-row x 320-dim bf16 embedding slab (zeros for pad dims / OOB word).
__device__ __forceinline__ void load_slab(__bf16 (*slab)[KPAD_W], const float* emb,
                                          const int (*sidx)[N_WORD], int w) {
    for (int e = threadIdx.x; e < 16 * KPAD_W; e += 256) {
        int row = e / KPAD_W;
        int d   = e - row * KPAD_W;
        float val = 0.f;
        if (w >= 0 && w < N_WORD && d < WORD_DIM) {
            int tok = sidx[row][w];
            val = emb[(size_t)tok * WORD_DIM + d];
        }
        slab[row][d] = f2bf(val);
    }
}

// ---------- kernel 1: weight prep (bf16 + WMMA B-fragment swizzle) ----------
// conv_w [400][300][3] -> wconv blocks [tap 3][kc 10][nt 25] of 512 bf16
// v      [400][200]    -> wv    blocks [kc 13][at 13]        of 512 bf16
__global__ void prep_kernel(const float* __restrict__ conv_w, const float* __restrict__ v,
                            __bf16* __restrict__ wconv, __bf16* __restrict__ wv) {
    int blk = blockIdx.x;
    for (int sub = 0; sub < 2; ++sub) {
        int e   = sub * 256 + threadIdx.x;   // 0..511 within block
        int l16 = e >> 4;                    // lane 0..31
        int j   = e & 15;
        int k   = ((l16 >> 4) << 3) + (j & 7) + ((j >> 3) << 4);     // K-local 0..31 within chunk
        int cl  = l16 & 15;                  // N-local 0..15 within tile
        if (blk < 750) {
            int nt = blk % 25; int rem = blk / 25;
            int kc = rem % 10; int t = rem / 10;
            int kk = kc * 32 + k;
            int c  = nt * 16 + cl;
            float val = (kk < WORD_DIM && c < CHANNEL)
                      ? conv_w[((size_t)c * WORD_DIM + kk) * 3 + t] : 0.f;
            wconv[(size_t)blk * 512 + e] = f2bf(val);
        } else {
            int b2 = blk - 750;
            int at = b2 % 13; int kc = b2 / 13;
            int kk = kc * 32 + k;
            int c  = at * 16 + cl;
            float val = (kk < CHANNEL && c < ATTN_DIM)
                      ? v[(size_t)kk * ATTN_DIM + c] : 0.f;
            wv[(size_t)b2 * 512 + e] = f2bf(val);
        }
    }
}

// ---------- kernel 2: conv GEMM + attention score ----------
__global__ void __launch_bounds__(256)
pass1_kernel(const int* __restrict__ idx, const float* __restrict__ emb,
             const float* __restrict__ conv_b, const float* __restrict__ vb,
             const float* __restrict__ q, const __bf16* __restrict__ wconv,
             const __bf16* __restrict__ wv, float* __restrict__ a_out) {
    __shared__ __align__(16) __bf16 s_emb[3][16][KPAD_W];
    __shared__ __align__(16) __bf16 s_c[16][CPAD];
    __shared__ float s_apart[16];
    __shared__ int   s_idx[16][N_WORD];
    __shared__ float s_cb[CHANNEL];
    __shared__ float s_vb[APAD];
    __shared__ float s_q[APAD];

    const int tid  = threadIdx.x;
    const int lane = tid & 31;
    const int wave = tid >> 5;
    const int b0   = blockIdx.x * 16;
    const int mbase = (lane & 16) ? 8 : 0;

    for (int e = tid; e < 16 * N_WORD; e += 256)
        (&s_idx[0][0])[e] = idx[(size_t)b0 * N_WORD + e];
    for (int e = tid; e < CHANNEL; e += 256) s_cb[e] = conv_b[e];
    for (int e = tid; e < APAD; e += 256) {
        s_vb[e] = (e < ATTN_DIM) ? vb[e] : 0.f;
        s_q[e]  = (e < ATTN_DIM) ? q[e]  : 0.f;
    }
    for (int e = tid; e < 256; e += 256)
        s_c[e >> 4][CHANNEL + (e & 15)] = f2bf(0.f);   // pad cols 400..415
    __syncthreads();

    load_slab(s_emb[2], emb, s_idx, -1);   // word -1 -> slot 2 (zeros)
    load_slab(s_emb[0], emb, s_idx, 0);

    for (int n = 0; n < N_WORD; ++n) {
        __syncthreads();                               // prev compute done
        load_slab(s_emb[(n + 1) % 3], emb, s_idx, n + 1);
        __syncthreads();

        // conv: C[16 x 400] = sum over 3 taps, K padded to 320 each
        for (int nt = wave; nt < 25; nt += 8) {
            v8f acc = {};
            for (int t = 0; t < 3; ++t) {
                const __bf16* abase = &s_emb[(n - 1 + t + 3) % 3][0][0];
                #pragma unroll
                for (int kc = 0; kc < 10; ++kc) {
                    v16bf A = load_a(abase, KPAD_W, kc);
                    v16bf Bf = load_b(wconv + (size_t)((t * 10 + kc) * 25 + nt) * 512);
                    acc = __builtin_amdgcn_wmma_f32_16x16x32_bf16(
                        false, A, false, Bf, (short)0, acc, false, false);
                }
            }
            int col = nt * 16 + (lane & 15);
            float bias = s_cb[col];
            #pragma unroll
            for (int r = 0; r < 8; ++r)
                s_c[mbase + r][col] = f2bf(acc[r] + bias);
        }
        if (tid < 16) s_apart[tid] = 0.f;
        __syncthreads();

        // attention: H = tanh(C @ v + vb); a = H @ q (row reduce via LDS atomics)
        for (int at = wave; at < 13; at += 8) {
            v8f acc = {};
            #pragma unroll
            for (int kc = 0; kc < 13; ++kc) {
                v16bf A  = load_a(&s_c[0][0], CPAD, kc);
                v16bf Bf = load_b(wv + (size_t)(kc * 13 + at) * 512);
                acc = __builtin_amdgcn_wmma_f32_16x16x32_bf16(
                    false, A, false, Bf, (short)0, acc, false, false);
            }
            int col  = at * 16 + (lane & 15);
            float qv = s_q[col], bb = s_vb[col];
            #pragma unroll
            for (int r = 0; r < 8; ++r) {
                float h = tanhf(acc[r] + bb);
                atomicAdd(&s_apart[mbase + r], h * qv);
            }
        }
        __syncthreads();
        if (tid < 16) a_out[(size_t)(b0 + tid) * N_WORD + n] = s_apart[tid];
    }
}

// ---------- kernel 3: softmax stats over the BATCH axis (per word position) ----------
__global__ void stats_kernel(const float* __restrict__ a, float* __restrict__ stats) {
    int n = blockIdx.x;                    // 0..29
    __shared__ float red[256];
    float mx = -3.4e38f;
    for (int b = threadIdx.x; b < B_TOTAL; b += 256)
        mx = fmaxf(mx, a[(size_t)b * N_WORD + n]);
    red[threadIdx.x] = mx; __syncthreads();
    for (int s = 128; s > 0; s >>= 1) {
        if (threadIdx.x < s) red[threadIdx.x] = fmaxf(red[threadIdx.x], red[threadIdx.x + s]);
        __syncthreads();
    }
    mx = red[0]; __syncthreads();
    float sm = 0.f;
    for (int b = threadIdx.x; b < B_TOTAL; b += 256)
        sm += __expf(a[(size_t)b * N_WORD + n] - mx);
    red[threadIdx.x] = sm; __syncthreads();
    for (int s = 128; s > 0; s >>= 1) {
        if (threadIdx.x < s) red[threadIdx.x] += red[threadIdx.x + s];
        __syncthreads();
    }
    if (threadIdx.x == 0) { stats[n * 2] = mx; stats[n * 2 + 1] = red[0]; }
}

// ---------- kernel 4: recompute conv, weighted pooling e = sum_n alpha*C ----------
__global__ void __launch_bounds__(256)
pass3_kernel(const int* __restrict__ idx, const float* __restrict__ emb,
             const float* __restrict__ conv_b, const float* __restrict__ a_in,
             const float* __restrict__ stats, const __bf16* __restrict__ wconv,
             float* __restrict__ out) {
    __shared__ __align__(16) __bf16 s_emb[3][16][KPAD_W];
    __shared__ int   s_idx[16][N_WORD];
    __shared__ float s_alpha[16][N_WORD];
    __shared__ float s_cb[CHANNEL];

    const int tid  = threadIdx.x;
    const int lane = tid & 31;
    const int wave = tid >> 5;
    const int b0   = blockIdx.x * 16;
    const int mbase = (lane & 16) ? 8 : 0;

    for (int e = tid; e < 16 * N_WORD; e += 256)
        (&s_idx[0][0])[e] = idx[(size_t)b0 * N_WORD + e];
    for (int e = tid; e < CHANNEL; e += 256) s_cb[e] = conv_b[e];
    __syncthreads();
    for (int e = tid; e < 16 * N_WORD; e += 256) {
        int m = e / N_WORD, nn = e - m * N_WORD;
        float mx = stats[nn * 2], sm = stats[nn * 2 + 1];
        s_alpha[m][nn] = __expf(a_in[(size_t)(b0 + m) * N_WORD + nn] - mx) / sm;
    }
    load_slab(s_emb[2], emb, s_idx, -1);
    load_slab(s_emb[0], emb, s_idx, 0);

    float eacc[4][8];
    #pragma unroll
    for (int i = 0; i < 4; ++i)
        #pragma unroll
        for (int r = 0; r < 8; ++r) eacc[i][r] = 0.f;

    for (int n = 0; n < N_WORD; ++n) {
        __syncthreads();
        load_slab(s_emb[(n + 1) % 3], emb, s_idx, n + 1);
        __syncthreads();
        int ti = 0;
        for (int nt = wave; nt < 25; nt += 8, ++ti) {
            v8f acc = {};
            for (int t = 0; t < 3; ++t) {
                const __bf16* abase = &s_emb[(n - 1 + t + 3) % 3][0][0];
                #pragma unroll
                for (int kc = 0; kc < 10; ++kc) {
                    v16bf A  = load_a(abase, KPAD_W, kc);
                    v16bf Bf = load_b(wconv + (size_t)((t * 10 + kc) * 25 + nt) * 512);
                    acc = __builtin_amdgcn_wmma_f32_16x16x32_bf16(
                        false, A, false, Bf, (short)0, acc, false, false);
                }
            }
            int col = nt * 16 + (lane & 15);
            float bias = s_cb[col];
            #pragma unroll
            for (int r = 0; r < 8; ++r)
                eacc[ti][r] += s_alpha[mbase + r][n] * (acc[r] + bias);
        }
    }
    {
        int ti = 0;
        for (int nt = wave; nt < 25; nt += 8, ++ti) {
            int col = nt * 16 + (lane & 15);
            #pragma unroll
            for (int r = 0; r < 8; ++r)
                out[(size_t)(b0 + mbase + r) * CHANNEL + col] = eacc[ti][r];
        }
    }
}

// ---------- host launch ----------
extern "C" void kernel_launch(void* const* d_in, const int* in_sizes, int n_in,
                              void* d_out, int out_size, void* d_ws, size_t ws_size,
                              hipStream_t stream) {
    const int*   idx    = (const int*)d_in[0];
    const float* emb    = (const float*)d_in[1];
    const float* conv_w = (const float*)d_in[2];
    const float* conv_b = (const float*)d_in[3];
    const float* v      = (const float*)d_in[4];
    const float* vb     = (const float*)d_in[5];
    const float* q      = (const float*)d_in[6];

    char* ws = (char*)d_ws;
    float*  a_buf = (float*)(ws + WS_A_OFF);
    __bf16* wconv = (__bf16*)(ws + WS_WC_OFF);
    __bf16* wv    = (__bf16*)(ws + WS_WV_OFF);
    float*  stats = (float*)(ws + WS_ST_OFF);

    prep_kernel<<<919, 256, 0, stream>>>(conv_w, v, wconv, wv);
    pass1_kernel<<<B_TOTAL / 16, 256, 0, stream>>>(idx, emb, conv_b, vb, q,
                                                   wconv, wv, a_buf);
    stats_kernel<<<N_WORD, 256, 0, stream>>>(a_buf, stats);
    pass3_kernel<<<B_TOTAL / 16, 256, 0, stream>>>(idx, emb, conv_b, a_buf, stats,
                                                   wconv, (float*)d_out);
}
